// HDCHLB_22041772163428
// MI455X (gfx1250) — matrix operand order repaired and checked by reading, not compiled, not run on hardware
//
#include <hip/hip_runtime.h>
#include <math.h>

#define DD 64          // embedding dim
#define TILE_M 16      // rows per wave tile
#define ETC 50000      // ET constant from reference

typedef __attribute__((ext_vector_type(16))) _Float16 v16h;
typedef __attribute__((ext_vector_type(8)))  float    v8f;

// ---------------------------------------------------------------------------
// B-fragment prep: convert W (64 x 64 f32, row-major) into WMMA B fragments.
// Fragment i = kt*4 + nt (8 frags for K=64), lane L:
//   N = nt*16 + (L&15),  K = kt*32 + (L>>4)*8 + e (e<8), +16+(e-8) (e>=8)
// Stored at Bfrag[(i*32 + L) * 16 .. +15]  -> two b128 loads at use.
// ---------------------------------------------------------------------------
__global__ void prep_bfrag(const float* __restrict__ W, _Float16* __restrict__ Bfrag) {
  const int slot = threadIdx.x;              // 256 slots: 2 kt * 4 nt * 32 lanes
  const int lane = slot & 31;
  const int nt = (slot >> 5) & 3;
  const int kt = slot >> 7;
  const int n  = (nt << 4) + (lane & 15);
  const int kb = (kt << 5) + ((lane >> 4) << 3);
  v16h b;
#pragma unroll
  for (int e = 0; e < 16; ++e) b[e] = (_Float16)W[(size_t)(kb + e + ((e >> 3) << 3)) * DD + n];
  *(v16h*)(Bfrag + (size_t)slot * 16) = b;
}

// ---------------------------------------------------------------------------
// A fragment: 16x32 f32 tile -> f16 fragment (row stride fixed at 64).
//   lanes 0-15 : M=lane,    K = kbase + {0..7}, kbase+16+{0..7}
//   lanes 16-31: M=lane-16, K = kbase+8+{0..7}, kbase+24+{0..7}
// ---------------------------------------------------------------------------
__device__ inline v16h cvt_a(float4 x0, float4 x1, float4 x2, float4 x3) {
  v16h a;
  a[0]  = (_Float16)x0.x; a[1]  = (_Float16)x0.y;
  a[2]  = (_Float16)x0.z; a[3]  = (_Float16)x0.w;
  a[4]  = (_Float16)x1.x; a[5]  = (_Float16)x1.y;
  a[6]  = (_Float16)x1.z; a[7]  = (_Float16)x1.w;
  a[8]  = (_Float16)x2.x; a[9]  = (_Float16)x2.y;
  a[10] = (_Float16)x2.z; a[11] = (_Float16)x2.w;
  a[12] = (_Float16)x3.x; a[13] = (_Float16)x3.y;
  a[14] = (_Float16)x3.z; a[15] = (_Float16)x3.w;
  return a;
}

__device__ inline v16h load_a_frag_fast(const float* __restrict__ A, int row0,
                                        int kbase, int lane) {
  const int m  = row0 + (lane & 15);
  const int kb = kbase + ((lane >> 4) << 3);
  const float4* s = (const float4*)(A + (size_t)m * DD + kb);
  return cvt_a(s[0], s[1], s[4], s[5]);
}

// Guarded path: clamp row so loads are always in-bounds (unconditional load +
// cndmask), zero contributions from out-of-range rows.
__device__ inline v16h load_a_frag_guard(const float* __restrict__ A, int row0, int nrows,
                                         int kbase, int lane) {
  int m = row0 + (lane & 15);
  const float z = (m < nrows) ? 1.f : 0.f;
  m = (m < nrows) ? m : (nrows - 1);
  const int kb = kbase + ((lane >> 4) << 3);
  const float4* s = (const float4*)(A + (size_t)m * DD + kb);
  float4 x0 = s[0], x1 = s[1], x2 = s[4], x3 = s[5];
  x0.x *= z; x0.y *= z; x0.z *= z; x0.w *= z;
  x1.x *= z; x1.y *= z; x1.z *= z; x1.w *= z;
  x2.x *= z; x2.y *= z; x2.z *= z; x2.w *= z;
  x3.x *= z; x3.y *= z; x3.z *= z; x3.w *= z;
  return cvt_a(x0, x1, x2, x3);
}

// ---------------------------------------------------------------------------
// Core: c[4] (16x64 f32 tile) += A(16x64 rows at row0) @ W(64x64, fragmented).
// All 8 B fragments preloaded up-front (one clause, wave-invariant addresses),
// both A fragments next, then 8 back-to-back WMMAs.
// ---------------------------------------------------------------------------
__device__ inline void mma_rows64(const float* __restrict__ A,
                                  const _Float16* __restrict__ Bfrag,
                                  int row0, int nrows, bool full, int lane, v8f c[4]) {
  v16h b[8];
#pragma unroll
  for (int i = 0; i < 8; ++i)
    b[i] = *(const v16h*)(Bfrag + (size_t)(i * 32 + lane) * 16);

  v16h a0, a1;
  if (full) {
    a0 = load_a_frag_fast(A, row0, 0, lane);
    a1 = load_a_frag_fast(A, row0, 32, lane);
  } else {
    a0 = load_a_frag_guard(A, row0, nrows, 0, lane);
    a1 = load_a_frag_guard(A, row0, nrows, 32, lane);
  }

#pragma unroll
  for (int nt = 0; nt < 4; ++nt) {
    c[nt] = __builtin_amdgcn_wmma_f32_16x16x32_f16(false, a0, false, b[nt],
                                                   (short)0, c[nt], false, false);
    c[nt] = __builtin_amdgcn_wmma_f32_16x16x32_f16(false, a1, false, b[4 + nt],
                                                   (short)0, c[nt], false, false);
  }
}

// ---------------------------------------------------------------------------
// GEMM: Out(nrows x 64) = A(nrows x 64) @ W(64 x 64)  [+= if accumulate]
// One wave = 16-row tile, 8 waves per block.
// ---------------------------------------------------------------------------
__global__ void wmma_gemm64(const float* __restrict__ A,
                            const _Float16* __restrict__ Bfrag,
                            float* __restrict__ Out, int nrows, int accumulate) {
  const int lane = threadIdx.x & 31;
  const int wave = threadIdx.x >> 5;
  const int tile = blockIdx.x * (blockDim.x >> 5) + wave;
  const int row0 = tile * TILE_M;
  if (row0 >= nrows) return;                    // wave-uniform
  const bool full = (row0 + TILE_M <= nrows);   // wave-uniform

  const int nlo = lane & 15;
  const int mofs = (lane >> 4) << 3;            // +0 or +8 row offset

  v8f c[4];
#pragma unroll
  for (int nt = 0; nt < 4; ++nt)
#pragma unroll
    for (int r = 0; r < 8; ++r) c[nt][r] = 0.f;

  if (accumulate) {
    if (full) {
#pragma unroll
      for (int nt = 0; nt < 4; ++nt) {
        const int n = (nt << 4) + nlo;
#pragma unroll
        for (int r = 0; r < 8; ++r)
          c[nt][r] = Out[(size_t)(row0 + r + mofs) * DD + n];
      }
    } else {
#pragma unroll
      for (int nt = 0; nt < 4; ++nt) {
        const int n = (nt << 4) + nlo;
#pragma unroll
        for (int r = 0; r < 8; ++r) {
          int m = row0 + r + mofs;
          const float z = (m < nrows) ? 1.f : 0.f;
          m = (m < nrows) ? m : (nrows - 1);
          c[nt][r] = Out[(size_t)m * DD + n] * z;
        }
      }
    }
  }

  mma_rows64(A, Bfrag, row0, nrows, full, lane, c);

  if (full) {
#pragma unroll
    for (int nt = 0; nt < 4; ++nt) {
      const int n = (nt << 4) + nlo;
#pragma unroll
      for (int r = 0; r < 8; ++r)
        Out[(size_t)(row0 + r + mofs) * DD + n] = c[nt][r];
    }
  } else {
#pragma unroll
    for (int nt = 0; nt < 4; ++nt) {
      const int n = (nt << 4) + nlo;
#pragma unroll
      for (int r = 0; r < 8; ++r) {
        const int m = row0 + r + mofs;
        if (m < nrows) Out[(size_t)m * DD + n] = c[nt][r];
      }
    }
  }
}

// ---------------------------------------------------------------------------
// Gate: Out = X * sigmoid(X @ W + b) ; X is nrows x 64, W pre-fragmented.
// ---------------------------------------------------------------------------
__global__ void wmma_gate(const float* __restrict__ X,
                          const _Float16* __restrict__ Bfrag,
                          const float* __restrict__ bias, float* __restrict__ Out,
                          int nrows) {
  const int lane = threadIdx.x & 31;
  const int wave = threadIdx.x >> 5;
  const int tile = blockIdx.x * (blockDim.x >> 5) + wave;
  const int row0 = tile * TILE_M;
  if (row0 >= nrows) return;
  const bool full = (row0 + TILE_M <= nrows);

  const int nlo = lane & 15;
  const int mofs = (lane >> 4) << 3;

  v8f c[4];
#pragma unroll
  for (int nt = 0; nt < 4; ++nt)
#pragma unroll
    for (int r = 0; r < 8; ++r) c[nt][r] = 0.f;

  mma_rows64(X, Bfrag, row0, nrows, full, lane, c);

  if (full) {
#pragma unroll
    for (int nt = 0; nt < 4; ++nt) {
      const int n = (nt << 4) + nlo;
      const float bn = bias[n];
#pragma unroll
      for (int r = 0; r < 8; ++r) {
        const size_t idx = (size_t)(row0 + r + mofs) * DD + n;
        const float t = c[nt][r] + bn;
        const float g = 1.f / (1.f + __expf(-t));
        Out[idx] = X[idx] * g;
      }
    }
  } else {
#pragma unroll
    for (int nt = 0; nt < 4; ++nt) {
      const int n = (nt << 4) + nlo;
      const float bn = bias[n];
#pragma unroll
      for (int r = 0; r < 8; ++r) {
        const int m = row0 + r + mofs;
        if (m < nrows) {
          const size_t idx = (size_t)m * DD + n;
          const float t = c[nt][r] + bn;
          const float g = 1.f / (1.f + __expf(-t));
          Out[idx] = X[idx] * g;
        }
      }
    }
  }
}

// ---------------------------------------------------------------------------
// SpMM scatter: Y[rows[i]] += vals[i] * X[cols[i]]   (rows of width 64)
// 16 threads per nnz, 4 columns each (b128 gather + 4x atomic_add_f32)
// ---------------------------------------------------------------------------
__global__ void spmm_scatter(const int* __restrict__ rows, const int* __restrict__ cols,
                             const float* __restrict__ vals, const float* __restrict__ X,
                             float* __restrict__ Y, int nnz) {
  const long t = (long)blockIdx.x * blockDim.x + threadIdx.x;
  const long i = t >> 4;
  if (i >= nnz) return;
  const int c4 = (int)(t & 15) << 2;
  const float v = vals[i];
  const float4 x = *(const float4*)(X + (size_t)cols[i] * DD + c4);
  float* y = Y + (size_t)rows[i] * DD + c4;
  atomicAdd(y + 0, v * x.x);
  atomicAdd(y + 1, v * x.y);
  atomicAdd(y + 2, v * x.z);
  atomicAdd(y + 3, v * x.w);
}

// ---------------------------------------------------------------------------
// Elementwise utilities
// ---------------------------------------------------------------------------
__global__ void k_fill0(float* __restrict__ p, size_t n) {
  const size_t i = (size_t)blockIdx.x * blockDim.x + threadIdx.x;
  if (i < n) p[i] = 0.f;
}
__global__ void k_copy(float* __restrict__ dst, const float* __restrict__ src, size_t n) {
  const size_t i = (size_t)blockIdx.x * blockDim.x + threadIdx.x;
  if (i < n) dst[i] = src[i];
}
// p = prop + p ; acc += p
__global__ void k_update(float* __restrict__ p, const float* __restrict__ prop,
                         float* __restrict__ acc, size_t n) {
  const size_t i = (size_t)blockIdx.x * blockDim.x + threadIdx.x;
  if (i < n) { const float t = p[i] + prop[i]; p[i] = t; acc[i] += t; }
}
__global__ void k_scale(float* __restrict__ dst, const float* __restrict__ src,
                        size_t n, float s) {
  const size_t i = (size_t)blockIdx.x * blockDim.x + threadIdx.x;
  if (i < n) dst[i] = src[i] * s;
}

// ---------------------------------------------------------------------------
// Masked mean pooling over user sequences (+ optional user-embedding add)
// grid = B blocks, 64 threads (one per column)
// ---------------------------------------------------------------------------
__global__ void pool_mean(const float* __restrict__ emb_pad,       // (P+1) x 64
                          const int* __restrict__ seq, const int* __restrict__ mask,
                          int T,
                          const float* __restrict__ user_out,      // U x 64 or null
                          const int* __restrict__ user_idx,
                          float* __restrict__ out) {
  const int b = blockIdx.x;
  const int d = threadIdx.x;
  float s = 0.f, cnt = 0.f;
  for (int t = 0; t < T; ++t) {
    const float m = (float)mask[(size_t)b * T + t];
    const int   p = seq[(size_t)b * T + t];
    cnt += m;
    s   += m * emb_pad[(size_t)p * DD + d];
  }
  const float denom = cnt < 1.f ? 1.f : cnt;
  float r = s / denom;
  if (user_out) r += user_out[(size_t)user_idx[b] * DD + d];
  out[(size_t)b * DD + d] = r;
}

// ---------------------------------------------------------------------------
// Host-side orchestration
// ---------------------------------------------------------------------------
static inline int cdiv_i(long n, int d) { return (int)((n + d - 1) / d); }

static void launch_gemm(const float* A, const float* W, _Float16* bfrag, float* O,
                        int nrows, int acc, hipStream_t s) {
  prep_bfrag<<<1, 256, 0, s>>>(W, bfrag);
  wmma_gemm64<<<cdiv_i(nrows, 128), 256, 0, s>>>(A, bfrag, O, nrows, acc);
}
static void launch_gate(const float* X, const float* W, _Float16* bfrag,
                        const float* bias, float* O, int nrows, hipStream_t s) {
  prep_bfrag<<<1, 256, 0, s>>>(W, bfrag);
  wmma_gate<<<cdiv_i(nrows, 128), 256, 0, s>>>(X, bfrag, bias, O, nrows);
}
static void launch_fill(float* p, size_t n, hipStream_t s) {
  k_fill0<<<cdiv_i((long)n, 256), 256, 0, s>>>(p, n);
}
static void launch_copy(float* d, const float* src, size_t n, hipStream_t s) {
  k_copy<<<cdiv_i((long)n, 256), 256, 0, s>>>(d, src, n);
}
static void launch_update(float* p, const float* prop, float* acc, size_t n, hipStream_t s) {
  k_update<<<cdiv_i((long)n, 256), 256, 0, s>>>(p, prop, acc, n);
}
static void launch_scale(float* d, const float* src, size_t n, float sc, hipStream_t s) {
  k_scale<<<cdiv_i((long)n, 256), 256, 0, s>>>(d, src, n, sc);
}
static void launch_spmm(const int* rows, const int* cols, const float* vals,
                        const float* X, float* Y, int nnz, hipStream_t s) {
  spmm_scatter<<<cdiv_i((long)nnz * 16, 256), 256, 0, s>>>(rows, cols, vals, X, Y, nnz);
}

// One hetero net: gated poi GNN with edge entity (user/region/category).
// On exit: bufP holds (P+1)x64 padded averaged poi output, bufEW holds averaged
// edge output (n_edge x 64).
static void hetero_net(const float* poi_emb, const float* gW, const float* gB,
                       const float* Wp, const float* We, const float* Wf,
                       const float* edge_emb, int n_edge,
                       const int* poi_idx, const int* edge_idx,
                       const float* vals_pe, const float* vals_ep, int nnz, int P,
                       float* bufP, float* bufAccP, float* bufPW,
                       float* bufE, float* bufAccE, float* bufEW, float* bufMsg,
                       float* bufFused, _Float16* bfrag, hipStream_t s) {
  const size_t nP = (size_t)P * DD, nE = (size_t)n_edge * DD;

  launch_gate(poi_emb, gW, bfrag, gB, bufP, P, s);
  launch_copy(bufAccP, bufP, nP, s);
  launch_copy(bufE, edge_emb, nE, s);
  launch_copy(bufAccE, edge_emb, nE, s);

  for (int layer = 0; layer < 2; ++layer) {
    launch_gemm(bufP, Wp, bfrag, bufPW, P, 0, s);                    // p @ Wp
    launch_fill(bufMsg, nE, s);
    launch_spmm(edge_idx, poi_idx, vals_pe, bufPW, bufMsg, nnz, s);  // poi_msg
    launch_gemm(bufE, We, bfrag, bufEW, n_edge, 0, s);               // e @ We
    launch_gemm(bufMsg, Wf,           bfrag, bufFused, n_edge, 0, s); // Wf top half
    launch_gemm(bufEW,  Wf + DD * DD, bfrag, bufFused, n_edge, 1, s); // Wf bottom (+=)
    launch_fill(bufPW, nP, s);                                       // prop (reuse bufPW)
    launch_spmm(poi_idx, edge_idx, vals_ep, bufFused, bufPW, nnz, s);
    launch_update(bufP, bufPW, bufAccP, nP, s);                      // p+=prop; accp+=p
    launch_update(bufE, bufFused, bufAccE, nE, s);                   // e+=fused; acce+=e
  }
  const float inv = 1.f / 3.f;
  launch_scale(bufP, bufAccP, nP, inv, s);   // averaged poi out
  launch_fill(bufP + nP, DD, s);             // zero pad row P
  launch_scale(bufEW, bufAccE, nE, inv, s);  // averaged edge out
}

extern "C" void kernel_launch(void* const* d_in, const int* in_sizes, int n_in,
                              void* d_out, int out_size, void* d_ws, size_t ws_size,
                              hipStream_t stream) {
  // -------- inputs (setup_inputs order) --------
  const int*   user_idx  = (const int*)  d_in[0];
  const int*   user_seq  = (const int*)  d_in[1];
  const int*   user_mask = (const int*)  d_in[2];
  const int*   col_poi   = (const int*)  d_in[3];
  const int*   col_user  = (const int*)  d_in[4];
  const float* col_pe    = (const float*)d_in[5];
  const float* col_ep    = (const float*)d_in[6];
  const int*   reg_poi   = (const int*)  d_in[7];
  const int*   reg_reg   = (const int*)  d_in[8];
  const float* reg_pe    = (const float*)d_in[9];
  const float* reg_ep    = (const float*)d_in[10];
  const int*   cat_poi   = (const int*)  d_in[11];
  const int*   cat_cat   = (const int*)  d_in[12];
  const float* cat_pe    = (const float*)d_in[13];
  const float* cat_ep    = (const float*)d_in[14];
  const int*   tr_poi    = (const int*)  d_in[15];
  const int*   tr_edge   = (const int*)  d_in[16];
  const float* tr_tar    = (const float*)d_in[17];
  const float* tr_src    = (const float*)d_in[18];
  const float* poi_emb   = (const float*)d_in[19];
  const float* user_emb  = (const float*)d_in[20];
  const float* region_emb= (const float*)d_in[21];
  const float* cat_emb   = (const float*)d_in[22];
  const float* wg_col    = (const float*)d_in[23];
  const float* bg_col    = (const float*)d_in[24];
  const float* wg_tr     = (const float*)d_in[25];
  const float* bg_tr     = (const float*)d_in[26];
  const float* wg_reg    = (const float*)d_in[27];
  const float* bg_reg    = (const float*)d_in[28];
  const float* wg_cat    = (const float*)d_in[29];
  const float* bg_cat    = (const float*)d_in[30];
  const float* col_Wp    = (const float*)d_in[31];
  const float* col_We    = (const float*)d_in[32];
  const float* col_Wf    = (const float*)d_in[33];
  const float* reg_Wp    = (const float*)d_in[34];
  const float* reg_We    = (const float*)d_in[35];
  const float* reg_Wf    = (const float*)d_in[36];
  const float* cat_Wp    = (const float*)d_in[37];
  const float* cat_We    = (const float*)d_in[38];
  const float* cat_Wf    = (const float*)d_in[39];

  const int P = in_sizes[19] / DD - 1;   // poi_emb is (P+1) x D
  const int U = in_sizes[20] / DD;
  const int R = in_sizes[21] / DD;
  const int C = in_sizes[22] / DD;
  const int B = in_sizes[0];
  const int T = in_sizes[1] / B;
  const int NNZ_COL = in_sizes[3];
  const int NNZ_REG = in_sizes[7];
  const int NNZ_CAT = in_sizes[11];
  const int NNZ_T   = in_sizes[15];

  // -------- workspace layout (floats) --------
  float* ws = (float*)d_ws;
  size_t off = 0;
  float* bufP    = ws + off; off += (size_t)(P + 1) * DD;   // poi state / padded out
  float* bufAccP = ws + off; off += (size_t)P * DD;
  float* bufPW   = ws + off; off += (size_t)P * DD;         // p@Wp, reused as prop
  const size_t EMAX = (size_t)(ETC > U ? ETC : U);
  float* bufE    = ws + off; off += EMAX * DD;              // edge state / msg_tar
  float* bufAccE = ws + off; off += (size_t)U * DD;
  float* bufEW   = ws + off; off += (size_t)U * DD;         // e@We, reused as avg edge out
  float* bufMsg  = ws + off; off += (size_t)U * DD;
  float* bufFused= ws + off; off += (size_t)U * DD;
  _Float16* bfrag= (_Float16*)(ws + off); off += 2048;      // 256 slots * 16 f16

  float* out = (float*)d_out;   // 4 x B x 64: [col_u, trans_u, reg_u, cat_u]
  const size_t nP = (size_t)P * DD;

  // ---- collaborative (poi <-> user) net, output row 0 ----
  hetero_net(poi_emb, wg_col, bg_col, col_Wp, col_We, col_Wf,
             user_emb, U, col_poi, col_user, col_pe, col_ep, NNZ_COL, P,
             bufP, bufAccP, bufPW, bufE, bufAccE, bufEW, bufMsg, bufFused, bfrag, stream);
  pool_mean<<<B, DD, 0, stream>>>(bufP, user_seq, user_mask, T, bufEW, user_idx,
                                  out + 0 * (size_t)B * DD);

  // ---- transition (directed) net, output row 1 ----
  launch_gate(poi_emb, wg_tr, bfrag, bg_tr, bufP, P, stream);
  launch_copy(bufAccP, bufP, nP, stream);
  for (int layer = 0; layer < 2; ++layer) {
    launch_fill(bufE, (size_t)ETC * DD, stream);
    launch_spmm(tr_edge, tr_poi, tr_tar, bufP, bufE, NNZ_T, stream);   // msg_tar (ET x 64)
    launch_fill(bufPW, nP, stream);
    launch_spmm(tr_poi, tr_edge, tr_src, bufE, bufPW, NNZ_T, stream);  // msg_src (P x 64)
    launch_update(bufP, bufPW, bufAccP, nP, stream);
  }
  launch_scale(bufP, bufAccP, nP, 1.f / 3.f, stream);
  launch_fill(bufP + nP, DD, stream);
  pool_mean<<<B, DD, 0, stream>>>(bufP, user_seq, user_mask, T, nullptr, nullptr,
                                  out + 1 * (size_t)B * DD);

  // ---- region net, output row 2 ----
  hetero_net(poi_emb, wg_reg, bg_reg, reg_Wp, reg_We, reg_Wf,
             region_emb, R, reg_poi, reg_reg, reg_pe, reg_ep, NNZ_REG, P,
             bufP, bufAccP, bufPW, bufE, bufAccE, bufEW, bufMsg, bufFused, bfrag, stream);
  pool_mean<<<B, DD, 0, stream>>>(bufP, user_seq, user_mask, T, nullptr, nullptr,
                                  out + 2 * (size_t)B * DD);

  // ---- category net, output row 3 ----
  hetero_net(poi_emb, wg_cat, bg_cat, cat_Wp, cat_We, cat_Wf,
             cat_emb, C, cat_poi, cat_cat, cat_pe, cat_ep, NNZ_CAT, P,
             bufP, bufAccP, bufPW, bufE, bufAccE, bufEW, bufMsg, bufFused, bfrag, stream);
  pool_mean<<<B, DD, 0, stream>>>(bufP, user_seq, user_mask, T, nullptr, nullptr,
                                  out + 3 * (size_t)B * DD);
}
